// SimpleHGNNLayer_23484881174650
// MI455X (gfx1250) — matrix operand order repaired and checked by reading, not compiled, not run on hardware
//
#include <hip/hip_runtime.h>

typedef __bf16 v16bf __attribute__((ext_vector_type(16)));
typedef float  v8f   __attribute__((ext_vector_type(8)));
typedef unsigned int u32x4 __attribute__((ext_vector_type(4)));
typedef int          i32x4 __attribute__((ext_vector_type(4)));
typedef int          i32x8 __attribute__((ext_vector_type(8)));

constexpr int Nn = 20000;   // nodes
constexpr int Ee = 8192;    // hyperedges
constexpr int Dd = 128;     // feature dim

static __device__ __forceinline__ __bf16 bf(float x) { return (__bf16)x; }

// ---------------------------------------------------------------------------
// Tensor Data Mover path (CDNA5). D# layout per CDNA5 ISA
// §8.3/8.4. Guarded so the file still compiles if the builtin is absent.
// ---------------------------------------------------------------------------
#if __has_builtin(__builtin_amdgcn_tensor_load_to_lds) && \
    __has_builtin(__builtin_amdgcn_s_wait_tensorcnt)
#define USE_TDM 1
#else
#define USE_TDM 0
#endif

#if USE_TDM
// LDS flat addresses carry the byte offset in the low 32 bits (ISA §10.2).
static __device__ __forceinline__ uint32_t ldsOff(const void* p) {
  return (uint32_t)(uintptr_t)p;
}

// 2-D tile load (data_size = 4B). tileH==1 gives a plain contiguous copy.
static __device__ __forceinline__ void tdm_load(
    const void* gptr, uint32_t lds, uint32_t tileW, uint32_t tileH,
    uint32_t tenW, uint32_t tenH, uint32_t strideElems) {
  const uint64_t ga = (uint64_t)(uintptr_t)gptr;
  u32x4 g0;
  g0[0] = 1u;                                    // count=1, user descriptor
  g0[1] = lds;                                   // lds_addr
  g0[2] = (uint32_t)ga;                          // global_addr[31:0]
  g0[3] = (uint32_t)((ga >> 32) & 0x01FFFFFFu)   // global_addr[56:32]
          | (2u << 30);                          // type = 2 ("image")
  i32x8 g1;
  g1[0] = (int)(2u << 16);                       // data_size code 2 = 4 bytes
  g1[1] = (int)((tenW & 0xFFFFu) << 16);         // tensor_dim0[15:0]
  g1[2] = (int)(((tenW >> 16) & 0xFFFFu) | ((tenH & 0xFFFFu) << 16));
  g1[3] = (int)(((tenH >> 16) & 0xFFFFu) | (tileW << 16));   // tile_dim0
  g1[4] = (int)(tileH & 0xFFFFu);                // tile_dim1 (tile_dim2 = 0)
  g1[5] = (int)strideElems;                      // tensor_dim0_stride[31:0]
  g1[6] = 0;
  g1[7] = 0;
  const i32x4 z4 = {0, 0, 0, 0};
#if defined(__clang_major__) && __clang_major__ >= 23
  const i32x8 z8 = {0, 0, 0, 0, 0, 0, 0, 0};
  __builtin_amdgcn_tensor_load_to_lds(g0, g1, z4, z4, z8, 0);
#else
  __builtin_amdgcn_tensor_load_to_lds(g0, g1, z4, z4, 0);
#endif
}
#endif  // USE_TDM

// ---------------------------------------------------------------------------
// Kernel A:  M[e,d] = De_inv[e] * sum_n H[n,e] * X[n,d]
// Grid: Ee/32 WGs x 256 threads. Wave w owns d-slice [16w,16w+16) and two
// e-subtiles. Double-buffered LDS; TDM (or vector loads) prefetch the next
// K-chunk while WMMAs consume the current one. De accumulated from the
// staged tile (one pass over H total).
// ---------------------------------------------------------------------------
__global__ __launch_bounds__(256) void hgnn_edge_gemm(
    const float* __restrict__ X, const float* __restrict__ H,
    float* __restrict__ M) {
  __shared__ float sH[2][32][32];    // [buf][n_local][e_local]   8 KB
  __shared__ float sX[2][32][128];   // [buf][n_local][d]        32 KB
  __shared__ float sRed[8][32];
  __shared__ float sDeInv[32];

  const int t    = threadIdx.x;
  const int wave = t >> 5;
  const int lane = t & 31;
  const int e0   = blockIdx.x * 32;

  auto issue = [&](int buf, int kb) {
    const int n0 = kb * 32;
#if USE_TDM
    if (wave == 0) {
      tdm_load(&H[(size_t)n0 * Ee + e0], ldsOff(&sH[buf][0][0]),
               /*tileW*/32, /*tileH*/32, /*tenW*/Ee, /*tenH*/Nn, /*stride*/Ee);
      tdm_load(&X[(size_t)n0 * Dd], ldsOff(&sX[buf][0][0]),
               /*tileW*/4096, /*tileH*/1, /*tenW*/(uint32_t)Nn * Dd, 1, 4096);
    }
#else
    const int hr = t >> 3, hc = (t & 7) * 4;     // 32x32 tile, one float4/thread
    *reinterpret_cast<float4*>(&sH[buf][hr][hc]) =
        *reinterpret_cast<const float4*>(&H[(size_t)(n0 + hr) * Ee + e0 + hc]);
    const int xr = t >> 3, xc = (t & 7) * 16;    // 32x128 tile, 4 float4/thread
    const float4* xp = reinterpret_cast<const float4*>(&X[(size_t)(n0 + xr) * Dd + xc]);
    float4*       sp = reinterpret_cast<float4*>(&sX[buf][xr][xc]);
    #pragma unroll
    for (int i = 0; i < 4; ++i) sp[i] = xp[i];
#endif
  };

  v8f acc0 = {}, acc1 = {};
  float dePart = 0.f;
  const int degC  = t & 31;
  const int degR0 = (t >> 5) * 4;

  constexpr int iters = Nn / 32;   // 625
  issue(0, 0);
  for (int kb = 0; kb < iters; ++kb) {
    const int cur = kb & 1;
#if USE_TDM
    if (wave == 0) __builtin_amdgcn_s_wait_tensorcnt(0);
#endif
    __syncthreads();                         // buffer `cur` ready for all waves
    if (kb + 1 < iters) issue(cur ^ 1, kb + 1);  // prefetch next chunk

    // hyperedge-degree accumulation from the staged tile
    #pragma unroll
    for (int j = 0; j < 4; ++j) dePart += sH[cur][degR0 + j][degC];

    // fragments per ISA 7.12.2 bf16 layouts
    const int row   = lane & 15;
    const int koffA = (lane >> 4) * 8;
    const int koffB = (lane >> 4) * 8;
    const int dcol  = wave * 16 + row;
    v16bf a0, a1, bb;
    #pragma unroll
    for (int v = 0; v < 8; ++v) {
      const int kk = koffA + ((v < 4) ? 2 * v : 16 + 2 * (v - 4));
      a0[2*v]   = bf(sH[cur][kk][row]);          // A = H^T : A[m=e][k=n]
      a0[2*v+1] = bf(sH[cur][kk + 1][row]);
      a1[2*v]   = bf(sH[cur][kk][16 + row]);
      a1[2*v+1] = bf(sH[cur][kk + 1][16 + row]);
      const int k2 = koffB + ((v < 4) ? 2 * v : 16 + 2 * (v - 4));
      bb[2*v]   = bf(sX[cur][k2][dcol]);         // B = X : B[k=n][n=d]
      bb[2*v+1] = bf(sX[cur][k2 + 1][dcol]);
    }
    acc0 = __builtin_amdgcn_wmma_f32_16x16x32_bf16(false, a0, false, bb,
                                                   (short)0, acc0, false, false);
    acc1 = __builtin_amdgcn_wmma_f32_16x16x32_bf16(false, a1, false, bb,
                                                   (short)0, acc1, false, false);
  }

  sRed[t >> 5][t & 31] = dePart;
  __syncthreads();
  if (t < 32) {
    float s = 0.f;
    #pragma unroll
    for (int j = 0; j < 8; ++j) s += sRed[j][t];
    sDeInv[t] = (s != 0.f) ? (1.f / s) : 0.f;
  }
  __syncthreads();

  const int mBase = (lane >> 4) * 8;   // C layout: VGPR r -> M=r / M=r+8
  const int c     = lane & 15;
  #pragma unroll
  for (int r = 0; r < 8; ++r) {
    const int m = r + mBase;
    M[(size_t)(e0 + m)      * Dd + wave * 16 + c] = acc0[r] * sDeInv[m];
    M[(size_t)(e0 + 16 + m) * Dd + wave * 16 + c] = acc1[r] * sDeInv[16 + m];
  }
}

// ---------------------------------------------------------------------------
// Kernel B:  out[n,:] = (Dv_inv[n] * sum_e H[n,e] * M[e,:]) @ W^T + b
// Grid: Nn/16 WGs x 256 threads. Same double-buffered TDM pipeline; Dv from
// the staged H tile; the 128-K projection reads W straight from L2.
// ---------------------------------------------------------------------------
__global__ __launch_bounds__(256) void hgnn_node_gemm(
    const float* __restrict__ H, const float* __restrict__ M,
    const float* __restrict__ W, const float* __restrict__ bias,
    float* __restrict__ out) {
  __shared__ float  sH[2][16][32];   // 4 KB
  __shared__ float  sM[2][32][128];  // 32 KB
  __shared__ __bf16 sY[16][128];     // scaled Y tile, 4 KB
  __shared__ float  sRed[16][16];
  __shared__ float  sDvInv[16];

  const int t    = threadIdx.x;
  const int wave = t >> 5;
  const int lane = t & 31;
  const int n0   = blockIdx.x * 16;

  auto issue = [&](int buf, int kb) {
    const int e0 = kb * 32;
#if USE_TDM
    if (wave == 0) {
      tdm_load(&H[(size_t)n0 * Ee + e0], ldsOff(&sH[buf][0][0]),
               /*tileW*/32, /*tileH*/16, /*tenW*/Ee, /*tenH*/Nn, /*stride*/Ee);
      tdm_load(&M[(size_t)e0 * Dd], ldsOff(&sM[buf][0][0]),
               /*tileW*/4096, /*tileH*/1, /*tenW*/(uint32_t)Ee * Dd, 1, 4096);
    }
#else
    if (t < 128) {                                 // 16x32 tile, one float4
      const int hr = t >> 3, hc = (t & 7) * 4;
      *reinterpret_cast<float4*>(&sH[buf][hr][hc]) =
          *reinterpret_cast<const float4*>(&H[(size_t)(n0 + hr) * Ee + e0 + hc]);
    }
    const int mr = t >> 3, mc = (t & 7) * 16;      // 32x128 tile, 4 float4
    const float4* mp = reinterpret_cast<const float4*>(&M[(size_t)(e0 + mr) * Dd + mc]);
    float4*       sp = reinterpret_cast<float4*>(&sM[buf][mr][mc]);
    #pragma unroll
    for (int i = 0; i < 4; ++i) sp[i] = mp[i];
#endif
  };

  v8f acc = {};
  float dvPart = 0.f;
  const int degR = t >> 4;
  const int degC = (t & 15) * 2;

  constexpr int iters = Ee / 32;   // 256
  issue(0, 0);
  for (int kb = 0; kb < iters; ++kb) {
    const int cur = kb & 1;
#if USE_TDM
    if (wave == 0) __builtin_amdgcn_s_wait_tensorcnt(0);
#endif
    __syncthreads();
    if (kb + 1 < iters) issue(cur ^ 1, kb + 1);

    // node-degree accumulation from the staged tile
    dvPart += sH[cur][degR][degC] + sH[cur][degR][degC + 1];

    const int row   = lane & 15;
    const int koffA = (lane >> 4) * 8;
    const int koffB = (lane >> 4) * 8;
    const int dcol  = wave * 16 + row;
    v16bf a, bb;
    #pragma unroll
    for (int v = 0; v < 8; ++v) {
      const int kk = koffA + ((v < 4) ? 2 * v : 16 + 2 * (v - 4));
      a[2*v]   = bf(sH[cur][row][kk]);        // A = H : A[m=n][k=e]
      a[2*v+1] = bf(sH[cur][row][kk + 1]);
      const int k2 = koffB + ((v < 4) ? 2 * v : 16 + 2 * (v - 4));
      bb[2*v]   = bf(sM[cur][k2][dcol]);      // B = M : B[k=e][n=d]
      bb[2*v+1] = bf(sM[cur][k2 + 1][dcol]);
    }
    acc = __builtin_amdgcn_wmma_f32_16x16x32_bf16(false, a, false, bb,
                                                  (short)0, acc, false, false);
  }

  sRed[degR][t & 15] = dvPart;
  __syncthreads();
  if (t < 16) {
    float s = 0.f;
    #pragma unroll
    for (int j = 0; j < 16; ++j) s += sRed[t][j];
    sDvInv[t] = (s != 0.f) ? (1.f / s) : 0.f;
  }
  __syncthreads();

  // scale by Dv_inv, spill Y tile to LDS as bf16
  const int mBase = (lane >> 4) * 8;
  const int c     = lane & 15;
  #pragma unroll
  for (int r = 0; r < 8; ++r) {
    const int m = r + mBase;
    sY[m][wave * 16 + c] = bf(acc[r] * sDvInv[m]);
  }
  __syncthreads();

  // phase 2: out = Y @ W^T + b  (K = 128 -> 4 WMMA steps, W from L2)
  v8f acc2 = {};
  const int row  = lane & 15;
  const int ocol = wave * 16 + row;     // output column o for the B fragment
  #pragma unroll
  for (int kc = 0; kc < 4; ++kc) {
    const int koffA = kc * 32 + (lane >> 4) * 8;
    const int koffB = kc * 32 + (lane >> 4) * 8;
    v16bf a, bb;
    #pragma unroll
    for (int v = 0; v < 8; ++v) {
      const int kk = koffA + ((v < 4) ? 2 * v : 16 + 2 * (v - 4));
      a[2*v]   = sY[row][kk];                  // A = Y : A[m=n][k=d]
      a[2*v+1] = sY[row][kk + 1];
      const int k2 = koffB + ((v < 4) ? 2 * v : 16 + 2 * (v - 4));
      const float2 wv =
          *reinterpret_cast<const float2*>(&W[(size_t)ocol * Dd + k2]);
      bb[2*v]   = bf(wv.x);                    // B = W^T : B[k=d][n=o]
      bb[2*v+1] = bf(wv.y);
    }
    acc2 = __builtin_amdgcn_wmma_f32_16x16x32_bf16(false, a, false, bb,
                                                   (short)0, acc2, false, false);
  }

  #pragma unroll
  for (int r = 0; r < 8; ++r) {
    const int m = r + mBase;
    out[(size_t)(n0 + m) * Dd + wave * 16 + c] = acc2[r] + bias[wave * 16 + c];
  }
}

// ---------------------------------------------------------------------------
extern "C" void kernel_launch(void* const* d_in, const int* in_sizes, int n_in,
                              void* d_out, int out_size, void* d_ws, size_t ws_size,
                              hipStream_t stream) {
  (void)in_sizes; (void)n_in; (void)out_size; (void)ws_size;
  const float* X = (const float*)d_in[0];   // [N, 128]
  const float* H = (const float*)d_in[1];   // [N, E]
  const float* W = (const float*)d_in[2];   // [128, 128]
  const float* b = (const float*)d_in[3];   // [128]
  float* out = (float*)d_out;               // [N, 128]
  float* M   = (float*)d_ws;                // [E, 128] fp32 scratch (4 MB)

  hgnn_edge_gemm<<<Ee / 32, 256, 0, stream>>>(X, H, M);
  hgnn_node_gemm<<<Nn / 16, 256, 0, stream>>>(H, M, W, b, out);
}
